// LowRankSparseAttention_84748294685525
// MI455X (gfx1250) — matrix operand order repaired and checked by reading, not compiled, not run on hardware
//
#include <hip/hip_runtime.h>
#include <hip/hip_bf16.h>

// ---- problem constants (match reference) ----
#define SQ      2048          // sequence length
#define DMODEL  768
#define NQKH    12
#define DQKH    64
#define NOVH    768
#define VKV     4
#define SKEY    (SQ + VKV)    // 2052 keys incl. virtual
#define KSTRIDE 2080          // padded KT row pitch (multiple of 32)
#define BATCH   2
#define INVSCALE 0.125f       // 1/sqrt(64)

typedef __attribute__((ext_vector_type(16))) _Float16 v16h;
typedef __attribute__((ext_vector_type(8)))  float    v8f;

static __device__ __forceinline__ v8f wmma16(v16h a, v16h b, v8f c) {
  // D = A(16x32 f16) * B(32x16 f16) + C(16x16 f32)
  return __builtin_amdgcn_wmma_f32_16x16x32_f16(false, a, false, b, (short)0, c,
                                                false, false);
}

// A-frag (16x32 f16, M=lane&15): two contiguous 16B chunks per lane.
// hw 0..7  -> K = (lane<16?0:8)+j ; hw 8..15 -> K = 16+(lane<16?0:8)+j
static __device__ __forceinline__ v16h a_frag_f16(const _Float16* base, int pitch) {
  const int lane = threadIdx.x & 31;
  const _Float16* p = base + (size_t)(lane & 15) * pitch + ((lane & 16) ? 8 : 0);
  union { v16h v; float4 q[2]; } u;
  u.q[0] = *(const float4*)p;
  u.q[1] = *(const float4*)(p + 16);
  return u.v;
}

// A-frag built from fp32 source (convert on the fly)
static __device__ __forceinline__ v16h a_frag_f32(const float* base, int pitch) {
  const int lane = threadIdx.x & 31;
  const float* p = base + (size_t)(lane & 15) * pitch + ((lane & 16) ? 8 : 0);
  union { v16h v; _Float16 h[16]; } u;
#pragma unroll
  for (int j = 0; j < 8; ++j) {
    u.h[j]     = (_Float16)p[j];
    u.h[8 + j] = (_Float16)p[16 + j];
  }
  return u.v;
}

// B-frag (32x16 f16): N = lane&15, K in table order, K pitch kp.
static __device__ __forceinline__ v16h b_frag_f16(const _Float16* p, int kp) {
  const int lane = threadIdx.x & 31;
  const _Float16* q = p + (lane & 15) + (size_t)((lane & 16) ? 8 : 0) * kp;
  union { v16h v; _Float16 h[16]; } u;
#pragma unroll
  for (int j = 0; j < 8; ++j) {
    u.h[j]     = q[(size_t)j * kp];
    u.h[8 + j] = q[(size_t)(16 + j) * kp];
  }
  return u.v;
}

// B-frag from fp32: N = lane&15, K in table order, K pitch kp (convert on the fly)
static __device__ __forceinline__ v16h b_frag_f32(const float* p, int kp) {
  const int lane = threadIdx.x & 31;
  const float* q = p + (lane & 15) + (size_t)((lane & 16) ? 8 : 0) * kp;
  union { v16h v; _Float16 h[16]; } u;
#pragma unroll
  for (int j = 0; j < 8; ++j) {
    u.h[j]     = (_Float16)q[(size_t)j * kp];
    u.h[8 + j] = (_Float16)q[(size_t)(16 + j) * kp];
  }
  return u.v;
}

// B-frag from fp32, N pitch stride, K contiguous in table order (W_V)
static __device__ __forceinline__ v16h b_frag_f32s(const float* p, int stride) {
  const int lane = threadIdx.x & 31;
  const float* q = p + (size_t)(lane & 15) * stride + ((lane & 16) ? 8 : 0);
  union { v16h v; _Float16 h[16]; } u;
#pragma unroll
  for (int j = 0; j < 8; ++j) {
    u.h[j]     = (_Float16)q[j];
    u.h[8 + j] = (_Float16)q[16 + j];
  }
  return u.v;
}

// ===================== Kernel 1: Q/K/V projection + bias + rotary ================
// Templated on PROJ (0=Q,1=K,2=V) so the k-loop is branch-free.
// grid = (4096/64, 12)  block = 32. One wave -> 64 rows x 64 cols, B-frag reuse x4.
template <int PROJ>
__global__ __launch_bounds__(32)
void lrsa_proj_kernel(const float* __restrict__ resid,
                      const float* __restrict__ W,     // W_Q / W_K / W_V
                      const float* __restrict__ bias,  // b_Q / b_K / b_V
                      const float* __restrict__ rsin, const float* __restrict__ rcos,
                      _Float16* __restrict__ Qbuf, _Float16* __restrict__ KT,
                      _Float16* __restrict__ Vbuf) {
  const int lane = threadIdx.x & 31;
  const int m0   = blockIdx.x * 64;        // global row in [0, B*S)
  const int n0   = blockIdx.y * 64;        // output-column tile base
  const int h    = n0 >> 6;

  v8f acc[4][4] = {};                      // [m-tile][n-slice]
  for (int kc = 0; kc < DMODEL; kc += 32) {
    v16h a[4];
#pragma unroll
    for (int mt = 0; mt < 4; ++mt)
      a[mt] = a_frag_f32(resid + (size_t)(m0 + 16 * mt) * DMODEL + kc, DMODEL);
    const int kg = kc + lane;              // B-frag: lane&15 is gemm-N column (kg unused)
#pragma unroll
    for (int t = 0; t < 4; ++t) {
      v16h bf;
      if (PROJ < 2)   // W_Q/W_K: [h, m, d] -> d = lane&15, m pitch DQKH
        bf = b_frag_f32(W + (size_t)h * (DMODEL * DQKH) + (size_t)kc * DQKH + 16 * t, DQKH);
      else            // W_V: [n, m] -> n pitch DMODEL, m contiguous
        bf = b_frag_f32s(W + (size_t)(n0 + 16 * t) * DMODEL + kc, DMODEL);
#pragma unroll
      for (int mt = 0; mt < 4; ++mt)
        acc[mt][t] = wmma16(a[mt], bf, acc[mt][t]);
    }
  }

  const int nlo = lane & 15;
  const int mh  = (lane & 16) ? 8 : 0;
#pragma unroll
  for (int mt = 0; mt < 4; ++mt) {
    const int mrow0 = m0 + 16 * mt;
    const int b     = mrow0 >> 11;                 // S = 2048
    const int s0    = mrow0 & (SQ - 1);
    if (PROJ < 2) {                                // Q or K: bias, rotary, store
      float xb[4][8];
#pragma unroll
      for (int t = 0; t < 4; ++t)
#pragma unroll
        for (int r = 0; r < 8; ++r)
          xb[t][r] = acc[mt][t][r] + bias[h * DQKH + 16 * t + nlo];
#pragma unroll
      for (int t = 0; t < 4; ++t) {
#pragma unroll
        for (int r = 0; r < 8; ++r) {
          const int d    = 16 * t + nlo;
          const int srow = s0 + r + mh;
          const float c  = rcos[srow * DQKH + d];
          const float sn = rsin[srow * DQKH + d];
          const float xp = xb[t ^ 2][r];           // partner at d +/- 32, same lane
          const float v  = (d < 32) ? (xb[t][r] * c - xp * sn)
                                    : (xb[t][r] * c + xp * sn);
          if (PROJ == 0)
            Qbuf[(((size_t)b * NQKH + h) * SQ + srow) * DQKH + d] = (_Float16)v;
          else
            KT[(((size_t)b * NQKH + h) * DQKH + d) * KSTRIDE + srow] = (_Float16)v;
        }
      }
    } else {                                       // V: bias, row-major store
#pragma unroll
      for (int t = 0; t < 4; ++t)
#pragma unroll
        for (int r = 0; r < 8; ++r) {
          const int n    = n0 + 16 * t + nlo;
          const int srow = s0 + r + mh;
          Vbuf[((size_t)b * SKEY + srow) * NOVH + n] =
              (_Float16)(acc[mt][t][r] + bias[n]);
        }
    }
  }
}

// ===================== Kernel 1b: fill virtual K columns / V rows ================
__global__ void lrsa_virt_kernel(const float* __restrict__ vk,
                                 const float* __restrict__ vv,
                                 _Float16* __restrict__ KT,
                                 _Float16* __restrict__ Vbuf) {
  const int idx    = blockIdx.x * blockDim.x + threadIdx.x;
  const int stride = blockDim.x * gridDim.x;
  for (int i = idx; i < VKV * NQKH * DQKH; i += stride) {   // [t,h,d]
    const int t = i / (NQKH * DQKH);
    const int h = (i / DQKH) % NQKH;
    const int d = i % DQKH;
    const _Float16 v = (_Float16)vk[i];
    for (int b = 0; b < BATCH; ++b)
      KT[(((size_t)b * NQKH + h) * DQKH + d) * KSTRIDE + SQ + t] = v;
  }
  for (int i = idx; i < VKV * NOVH; i += stride) {          // [t,n]
    const int t = i / NOVH;
    const int n = i % NOVH;
    const _Float16 v = (_Float16)vv[i];
    for (int b = 0; b < BATCH; ++b)
      Vbuf[((size_t)b * SKEY + SQ + t) * NOVH + n] = v;
  }
}

// ===================== Kernel 2: flash attention per (q-tile, head, batch) =======
// grid = (S/16, 12, 2)  block = 32. Online softmax, key tiles of 32.
__global__ __launch_bounds__(32)
void lrsa_attn_kernel(const _Float16* __restrict__ Qbuf,
                      const _Float16* __restrict__ KT,
                      const _Float16* __restrict__ Vbuf,
                      _Float16* __restrict__ Zbuf) {
  __shared__ __align__(16) _Float16 Plds[16 * 32];
  const int lane = threadIdx.x & 31;
  const int q0   = blockIdx.x * 16;
  const int h    = blockIdx.y;
  const int b    = blockIdx.z;
  const int nlo  = lane & 15;
  const int mh   = (lane & 16) ? 8 : 0;

  const _Float16* qbase = Qbuf + (((size_t)b * NQKH + h) * SQ + q0) * DQKH;
  const v16h aq0 = a_frag_f16(qbase, DQKH);        // dqk 0..31
  const v16h aq1 = a_frag_f16(qbase + 32, DQKH);   // dqk 32..63
  const _Float16* ktbase = KT + ((size_t)b * NQKH + h) * (size_t)(DQKH * KSTRIDE);
  const _Float16* vbase  = Vbuf + (size_t)b * SKEY * NOVH + h * DQKH;

  v8f zacc[4] = {};
  float mrow[8], lrow[8];
#pragma unroll
  for (int r = 0; r < 8; ++r) { mrow[r] = -1e30f; lrow[r] = 0.f; }

  const int jmax   = q0 + 15 + VKV;                // last visible key (incl.)
  const int ntiles = jmax / 32 + 1;

  for (int kt = 0; kt < ntiles; ++kt) {
    const int kbase = kt * 32;
    // prefetch next tile's K rows and V rows toward the WGP (global_prefetch_b8)
    if (kt + 1 < ntiles) {
      __builtin_prefetch(ktbase + (size_t)lane * KSTRIDE + kbase + 32, 0, 3);
      __builtin_prefetch(ktbase + (size_t)(32 + lane) * KSTRIDE + kbase + 32, 0, 3);
      __builtin_prefetch(vbase + (size_t)(kbase + 32 + lane) * NOVH, 0, 3);
    }
    // ---- scores: two 16x16 C tiles over 32 keys ----
    v8f s[2] = {};
    const _Float16* kr0 = ktbase + kbase;                                 // dqk 0..31
    const _Float16* kr1 = ktbase + (size_t)32 * KSTRIDE + kbase;          // dqk 32..63
#pragma unroll
    for (int t = 0; t < 2; ++t) {
      s[t] = wmma16(aq0, b_frag_f16(kr0 + 16 * t, KSTRIDE), s[t]);
      s[t] = wmma16(aq1, b_frag_f16(kr1 + 16 * t, KSTRIDE), s[t]);
    }
    // ---- scale + unified mask: key j visible iff j <= i+VKV and j < S+VKV ----
    float p[2][8], tmax[8];
#pragma unroll
    for (int r = 0; r < 8; ++r) {
      const int i  = q0 + r + mh;
      const int j0 = kbase + nlo, j1 = kbase + 16 + nlo;
      float v0 = s[0][r] * INVSCALE;
      float v1 = s[1][r] * INVSCALE;
      if (j0 > i + VKV || j0 >= SKEY) v0 = -1e30f;
      if (j1 > i + VKV || j1 >= SKEY) v1 = -1e30f;
      p[0][r] = v0; p[1][r] = v1;
      tmax[r] = fmaxf(v0, v1);
    }
    // row-max across the 16-lane half (rows differ between halves)
#pragma unroll
    for (int r = 0; r < 8; ++r)
#pragma unroll
      for (int x = 1; x < 16; x <<= 1)
        tmax[r] = fmaxf(tmax[r], __shfl_xor(tmax[r], x));
    float scl[8];
#pragma unroll
    for (int r = 0; r < 8; ++r) {
      const float mn = fmaxf(mrow[r], tmax[r]);
      scl[r]  = __expf(mrow[r] - mn);
      mrow[r] = mn;
      p[0][r] = __expf(p[0][r] - mn);
      p[1][r] = __expf(p[1][r] - mn);
    }
#pragma unroll
    for (int r = 0; r < 8; ++r) {
      float ts = p[0][r] + p[1][r];
#pragma unroll
      for (int x = 1; x < 16; x <<= 1) ts += __shfl_xor(ts, x);
      lrow[r] = lrow[r] * scl[r] + ts;
    }
    // ---- relayout P (C-tile -> A-frag) through LDS; rescale Z ----
    __syncthreads();
#pragma unroll
    for (int t = 0; t < 2; ++t)
#pragma unroll
      for (int r = 0; r < 8; ++r)
        Plds[(r + mh) * 32 + 16 * t + nlo] = (_Float16)p[t][r];
#pragma unroll
    for (int t = 0; t < 4; ++t)
#pragma unroll
      for (int r = 0; r < 8; ++r)
        zacc[t][r] *= scl[r];
    __syncthreads();
    const v16h pa = a_frag_f16(Plds, 32);
    // ---- Z += P(16x32) * V(32x64); lane&15 = V column, lane half picks keys ----
    const int vkh = (lane & 16) ? 8 : 0;
#pragma unroll
    for (int ns = 0; ns < 4; ++ns) {
      union { v16h v; _Float16 h[16]; } ub;
#pragma unroll
      for (int j = 0; j < 8; ++j) {
        int r0 = kbase + vkh + j, r1 = kbase + 16 + vkh + j;
        if (r0 > SKEY - 1) r0 = SKEY - 1;            // masked keys: avoid OOB
        if (r1 > SKEY - 1) r1 = SKEY - 1;
        ub.h[j]     = vbase[(size_t)r0 * NOVH + 16 * ns + nlo];
        ub.h[8 + j] = vbase[(size_t)r1 * NOVH + 16 * ns + nlo];
      }
      zacc[ns] = wmma16(pa, ub.v, zacc[ns]);
    }
  }
  // ---- normalize and store Z (f16, row-major [B,S,768]) ----
#pragma unroll
  for (int t = 0; t < 4; ++t)
#pragma unroll
    for (int r = 0; r < 8; ++r) {
      const int srow = q0 + r + mh;
      const int n    = h * DQKH + 16 * t + nlo;
      Zbuf[((size_t)b * SQ + srow) * NOVH + n] = (_Float16)(zacc[t][r] / lrow[r]);
    }
}

// ===================== Kernel 3: output projection out = Z @ W_O =================
// grid = (4096/64, 12)  block = 32.  W_O[h,0,:] is row-major [768 x 768].
__global__ __launch_bounds__(32)
void lrsa_outproj_kernel(const _Float16* __restrict__ Zbuf,
                         const float* __restrict__ W_O,
                         float* __restrict__ out) {
  const int lane = threadIdx.x & 31;
  const int m0   = blockIdx.x * 64;
  const int n0   = blockIdx.y * 64;
  v8f acc[4][4] = {};
  for (int kc = 0; kc < NOVH; kc += 32) {
    v16h a[4];
#pragma unroll
    for (int mt = 0; mt < 4; ++mt)
      a[mt] = a_frag_f16(Zbuf + (size_t)(m0 + 16 * mt) * NOVH + kc, NOVH);
    const int kg = kc + lane;
#pragma unroll
    for (int t = 0; t < 4; ++t) {
      const v16h bf = b_frag_f32(W_O + (size_t)kc * DMODEL + n0 + 16 * t, DMODEL);
#pragma unroll
      for (int mt = 0; mt < 4; ++mt)
        acc[mt][t] = wmma16(a[mt], bf, acc[mt][t]);
    }
  }
  const int nlo = lane & 15, mh = (lane & 16) ? 8 : 0;
#pragma unroll
  for (int mt = 0; mt < 4; ++mt)
#pragma unroll
    for (int t = 0; t < 4; ++t)
#pragma unroll
      for (int r = 0; r < 8; ++r)
        out[(size_t)(m0 + 16 * mt + r + mh) * DMODEL + n0 + 16 * t + nlo] =
            acc[mt][t][r];
}

// ===================== host launcher =============================================
extern "C" void kernel_launch(void* const* d_in, const int* in_sizes, int n_in,
                              void* d_out, int out_size, void* d_ws, size_t ws_size,
                              hipStream_t stream) {
  const float* resid = (const float*)d_in[0];
  const float* W_Q   = (const float*)d_in[1];
  const float* W_K   = (const float*)d_in[2];
  const float* W_V   = (const float*)d_in[3];
  const float* W_O   = (const float*)d_in[4];
  const float* b_Q   = (const float*)d_in[5];
  const float* b_K   = (const float*)d_in[6];
  const float* b_V   = (const float*)d_in[7];
  const float* vk    = (const float*)d_in[8];
  const float* vv    = (const float*)d_in[9];
  const float* rsin  = (const float*)d_in[10];
  const float* rcos  = (const float*)d_in[11];
  float* out = (float*)d_out;

  // workspace carve-up (f16): Q | KT | V | Z  (~25 MB total)
  _Float16* Qbuf = (_Float16*)d_ws;
  _Float16* KT   = Qbuf + (size_t)BATCH * NQKH * SQ * DQKH;
  _Float16* Vbuf = KT   + (size_t)BATCH * NQKH * DQKH * KSTRIDE;
  _Float16* Zbuf = Vbuf + (size_t)BATCH * SKEY * NOVH;

  const dim3 gproj((BATCH * SQ) / 64, DMODEL / 64);
  lrsa_proj_kernel<0><<<gproj, 32, 0, stream>>>(resid, W_Q, b_Q, rsin, rcos,
                                                Qbuf, KT, Vbuf);
  lrsa_proj_kernel<1><<<gproj, 32, 0, stream>>>(resid, W_K, b_K, rsin, rcos,
                                                Qbuf, KT, Vbuf);
  lrsa_proj_kernel<2><<<gproj, 32, 0, stream>>>(resid, W_V, b_V, rsin, rcos,
                                                Qbuf, KT, Vbuf);
  lrsa_virt_kernel<<<12, 256, 0, stream>>>(vk, vv, KT, Vbuf);
  lrsa_attn_kernel<<<dim3(SQ / 16, NQKH, BATCH), 32, 0, stream>>>(Qbuf, KT, Vbuf,
                                                                  Zbuf);
  lrsa_outproj_kernel<<<dim3((BATCH * SQ) / 64, DMODEL / 64), 32, 0, stream>>>(
      Zbuf, W_O, out);
}
